// CAPMemory_26680336843534
// MI455X (gfx1250) — matrix operand order repaired and checked by reading, not compiled, not run on hardware
//
#include <hip/hip_runtime.h>
#include <hip/hip_bf16.h>

typedef __attribute__((ext_vector_type(16))) __bf16    v16bf;
typedef __attribute__((ext_vector_type(2)))  __bf16    v2bf;
typedef __attribute__((ext_vector_type(8)))  float     v8f;
typedef __attribute__((ext_vector_type(8)))  int       v8i;
typedef __attribute__((ext_vector_type(4)))  int       v4i;
typedef __attribute__((ext_vector_type(4)))  unsigned  v4u;
typedef __attribute__((ext_vector_type(4)))  float     v4f;

#define B_SZ   256
#define D_SZ   2048
#define NBANK  16384       // 8 cams * 2048
#define CCAM   8
#define PIDS   2048
#define KTOP   50
#define NSLICE 32
#define SLICE  (NBANK / NSLICE)   // 512
#define INV_T  (1.0f / 0.07f)
#define KPAIRS 1024               // D_SZ / 2 packed bf16 pairs per row
#define LROW   20                 // LDS row stride in dwords (16 data + 4 pad)

union BfPack { v8i i; v16bf b; };
union BLoad  { v4u q[2]; v16bf b; };

__device__ __forceinline__ unsigned pk_bf16(float lo, float hi) {
#if __has_builtin(__builtin_amdgcn_cvt_pk_bf16_f32)
    v2bf p = __builtin_amdgcn_cvt_pk_bf16_f32(lo, hi);
    return __builtin_bit_cast(unsigned, p);
#else
    __bf16 l = (__bf16)lo;
    __bf16 h = (__bf16)hi;
    unsigned short ul = __builtin_bit_cast(unsigned short, l);
    unsigned short uh = __builtin_bit_cast(unsigned short, h);
    return ((unsigned)uh << 16) | (unsigned)ul;
#endif
}

// ---------------------------------------------------------------------------
// Kernel 1: L2-normalize each input row, pack to bf16 pairs, store b-major
// xTb[b][kpair] (coalesced stores; feeds the TDM's padded LDS tiles).
// ---------------------------------------------------------------------------
__global__ void __launch_bounds__(256) capNormPack(const float* __restrict__ x,
                                                   unsigned* __restrict__ xTb) {
    const int b = blockIdx.x;
    const int t = threadIdx.x;
    __shared__ float red[256];
    const float* row = x + (size_t)b * D_SZ;
    float s = 0.f;
#pragma unroll
    for (int j = 0; j < 8; ++j) { float v = row[t + 256 * j]; s += v * v; }
    red[t] = s;
    __syncthreads();
    for (int off = 128; off > 0; off >>= 1) {
        if (t < off) red[t] += red[t + off];
        __syncthreads();
    }
    const float inv = rsqrtf(fmaxf(red[0], 1e-24f));
#pragma unroll
    for (int j = 0; j < 4; ++j) {
        int kp = t + 256 * j;                       // kpair in [0,1024)
        float lo = row[2 * kp]     * inv;
        float hi = row[2 * kp + 1] * inv;
        xTb[(size_t)b * KPAIRS + kp] = pk_bf16(lo, hi);
    }
}

// ---------------------------------------------------------------------------
// TDM: async-load one K-chunk of xTb into LDS with row padding.
// Global tensor: 256 rows x 1024 dwords (stride 1024); tile = 16 x 256.
// pad_enable inserts 4 dwords after every 16 stored dwords -> LDS row
// stride 20 dwords (16B aligned, conflict-free ds_load_b128).
// ---------------------------------------------------------------------------
__device__ __forceinline__ void tdm_load_chunk(const unsigned* gsrc, unsigned lds_off) {
    unsigned long long ga = (unsigned long long)gsrc;
    v4u g0 = { 1u,                                   // count=1, user mode
               lds_off,                              // lds_addr (bytes)
               (unsigned)(ga & 0xFFFFFFFFu),         // global_addr[31:0]
               (unsigned)((ga >> 32) & 0x1FFFFFFu) | (2u << 30) }; // addr[56:32] | type=2
    v8i g1 = { (int)((2u << 16) | (1u << 20) | (3u << 22) | (3u << 25)),
               //    data_size=4B  pad_enable  pad_int=16dw  pad_amt=4dw
               (int)(1024u << 16),     // tensor_dim0[15:0] = 1024
               (int)(256u  << 16),     // tensor_dim1[15:0] = 256 rows
               (int)(16u   << 16),     // tile_dim0 = 16 dwords
               256,                    // tile_dim1 = 256 rows
               1024,                   // tensor_dim0_stride = 1024 dwords
               0, 0 };
    v4i gz4 = { 0, 0, 0, 0 };
    v8i gz8 = { 0, 0, 0, 0, 0, 0, 0, 0 };
    __builtin_amdgcn_tensor_load_to_lds(g0, g1, gz4, gz4, gz8, 0);
}

struct AQ { v4f q0, q1, q2, q3; };

__device__ __forceinline__ AQ load_a(const float* ap) {
    AQ a;
    a.q0 = *(const v4f*)(ap);
    a.q1 = *(const v4f*)(ap + 4);
    a.q2 = *(const v4f*)(ap + 16);
    a.q3 = *(const v4f*)(ap + 20);
    return a;
}

__device__ __forceinline__ void load_b(BLoad& d, const unsigned* bp) {
    d.q[0] = *(const v4u*)(bp);
    d.q[1] = *(const v4u*)(bp + 8);
}

// ---------------------------------------------------------------------------
// Kernel 2: logitsT[n][b] = xhat[b] . tempV[n]   via v_wmma_f32_16x16x32_bf16
// 128 WGs x 8 waves; each wave: 16 tempV rows x all 256 batch rows.
// tempV f32 read exactly once, cvt to bf16 in-register (A operand, prefetched
// one K-chunk ahead); x staged in LDS by TDM (double buffer); B operand
// double-buffered in registers so ds loads overlap WMMA issue.
// ---------------------------------------------------------------------------
__global__ void __launch_bounds__(256) capGemm(const float* __restrict__ tempV,
                                               const unsigned* __restrict__ xTb,
                                               float* __restrict__ logitsT) {
    __shared__ __attribute__((aligned(16))) unsigned sxt[2][B_SZ * LROW]; // 2 x 20 KB
    const int t    = threadIdx.x;
    const int wave = t >> 5;
    const int lane = t & 31;
    const int l15  = lane & 15;
    const int hi   = lane >> 4;                     // 0/1 half-wave
    const int nBase = blockIdx.x * 128 + wave * 16;
    const int nRow  = nBase + l15;

    v8f zero = {0.f, 0.f, 0.f, 0.f, 0.f, 0.f, 0.f, 0.f};
    v8f acc[16];
#pragma unroll
    for (int i = 0; i < 16; ++i) acc[i] = zero;

    const unsigned ldsBase = (unsigned)(unsigned long long)(&sxt[0][0]);
    const float*   aRow    = tempV + (size_t)nRow * D_SZ + hi * 8;

    AQ aq = load_a(aRow);                           // chunk 0 A prefetch
    if (wave == 0) {
        tdm_load_chunk(xTb, ldsBase);
        __builtin_amdgcn_s_wait_tensorcnt(0);
    }
    __syncthreads();

    for (int kc = 0; kc < 64; ++kc) {
        if (wave == 0 && kc + 1 < 64) {             // prefetch next LDS chunk
            tdm_load_chunk(xTb + (kc + 1) * 16,
                           ldsBase + (unsigned)(((kc + 1) & 1) * B_SZ * LROW * 4));
        }
        // prefetch next chunk's A quads (clamped redundant load on last iter)
        const int kcn = (kc + 1 < 64) ? (kc + 1) : 63;
        AQ aqn = load_a(aRow + kcn * 32);

        // convert current A quads -> bf16 operand
        BfPack A;
        A.i[0] = pk_bf16(aq.q0.x, aq.q0.y); A.i[1] = pk_bf16(aq.q0.z, aq.q0.w);
        A.i[2] = pk_bf16(aq.q1.x, aq.q1.y); A.i[3] = pk_bf16(aq.q1.z, aq.q1.w);
        A.i[4] = pk_bf16(aq.q2.x, aq.q2.y); A.i[5] = pk_bf16(aq.q2.z, aq.q2.w);
        A.i[6] = pk_bf16(aq.q3.x, aq.q3.y); A.i[7] = pk_bf16(aq.q3.z, aq.q3.w);

        const unsigned* lb = &sxt[kc & 1][0];
        const unsigned* bp0 = lb + l15 * LROW + hi * 4;
        BLoad Bb[2];
        load_b(Bb[0], bp0);                         // tile 0
#pragma unroll
        for (int bt = 0; bt < 16; ++bt) {
            if (bt + 1 < 16)                        // prefetch next tile
                load_b(Bb[(bt + 1) & 1], bp0 + (bt + 1) * 16 * LROW);
            acc[bt] = __builtin_amdgcn_wmma_f32_16x16x32_bf16(
                false, A.b, false, Bb[bt & 1].b, (short)0, acc[bt], false, false);
        }
        if (wave == 0) __builtin_amdgcn_s_wait_tensorcnt(0);
        __syncthreads();
        aq = aqn;
    }

    // D layout: VGPR r, lane l -> row M = r + 8*(l>>4), col N = l&15
#pragma unroll
    for (int bt = 0; bt < 16; ++bt)
#pragma unroll
        for (int r = 0; r < 8; ++r)
            logitsT[(size_t)(nBase + r + 8 * hi) * B_SZ + bt * 16 + l15] = acc[bt][r];
}

// ---------------------------------------------------------------------------
// Kernel 3: per n-slice partials. thread = batch row b; reads logitsT[n][b]
// coalesced.  Emits: own-cam online (max, sumexp) of v/T, slice top-50
// non-positive values, and the 8 positive ("ori") logits.
// ---------------------------------------------------------------------------
__global__ void __launch_bounds__(256) capPartials(const float* __restrict__ logitsT,
                                                   const int* __restrict__ labels,
                                                   const int* __restrict__ cams,
                                                   float* __restrict__ pmax,
                                                   float* __restrict__ psum,
                                                   float* __restrict__ ptop,
                                                   float* __restrict__ oriv) {
    const int s = blockIdx.x;
    const int t = threadIdx.x;                      // batch row
    const int lab = labels[t];
    const int cam = cams[t];
    const int n0  = s * SLICE;
    const int cSlice = n0 >> 11;                    // slice lies in one cam bank
    const bool inBank = (cSlice == cam);

    __shared__ float topl[256 * KTOP];
    float* my = &topl[t * KTOP];
    int cnt = 0, imin = 0;
    float vmin = 1e30f;
    float m = -1e30f, ssum = 0.f;

    for (int i = 0; i < SLICE; ++i) {
        int n = n0 + i;
        float v = logitsT[(size_t)n * B_SZ + t];
        bool isPos = ((n & (PIDS - 1)) == lab);
        if (isPos) oriv[t * CCAM + cSlice] = v;
        if (inBank) {                               // intra-cam LSE includes positive
            float z = v * INV_T;
            if (z > m) { ssum = ssum * __expf(m - z) + 1.f; m = z; }
            else        ssum += __expf(z - m);
        }
        if (!isPos) {                               // hard-negative candidate
            if (cnt < KTOP) {
                my[cnt] = v;
                if (v < vmin) { vmin = v; imin = cnt; }
                ++cnt;
            } else if (v > vmin) {
                my[imin] = v;
                vmin = my[0]; imin = 0;
                for (int j = 1; j < KTOP; ++j) {
                    float q = my[j];
                    if (q < vmin) { vmin = q; imin = j; }
                }
            }
        }
    }
    pmax[s * B_SZ + t] = m;
    psum[s * B_SZ + t] = ssum;
    float* dst = &ptop[(size_t)(s * B_SZ + t) * KTOP];
    for (int j = 0; j < KTOP; ++j) dst[j] = (j < cnt) ? my[j] : -1e30f;
}

// ---------------------------------------------------------------------------
// Kernel 4: merge partials -> ce[b], loss_k[b]; segment-mean over cameras.
// ---------------------------------------------------------------------------
__global__ void __launch_bounds__(256) capFinal(const float* __restrict__ pmax,
                                                const float* __restrict__ psum,
                                                const float* __restrict__ ptop,
                                                const float* __restrict__ oriv,
                                                const int* __restrict__ cams,
                                                float* __restrict__ out) {
    const int t = threadIdx.x;                      // batch row
    const int cam = cams[t];

    // ---- merge intra-cam LSE ----
    float m = -1e30f;
    for (int s = 0; s < NSLICE; ++s) m = fmaxf(m, pmax[s * B_SZ + t]);
    float sum = 0.f;
    for (int s = 0; s < NSLICE; ++s)
        sum += psum[s * B_SZ + t] * __expf(pmax[s * B_SZ + t] - m);
    float lse_i = m + __logf(fmaxf(sum, 1e-30f));
    float ce = lse_i - oriv[t * CCAM + cam] * INV_T;

    // ---- merge top-50 over all slices ----
    __shared__ float topl[256 * KTOP];
    float* my = &topl[t * KTOP];
    int cnt = 0, imin = 0;
    float vmin = 1e30f;
    for (int s = 0; s < NSLICE; ++s) {
        const float* src = &ptop[(size_t)(s * B_SZ + t) * KTOP];
        for (int j = 0; j < KTOP; ++j) {
            float v = src[j];
            if (cnt < KTOP) {
                my[cnt] = v;
                if (v < vmin) { vmin = v; imin = cnt; }
                ++cnt;
            } else if (v > vmin) {
                my[imin] = v;
                vmin = my[0]; imin = 0;
                for (int q = 1; q < KTOP; ++q)
                    if (my[q] < vmin) { vmin = my[q]; imin = q; }
            }
        }
    }

    // ---- inter-cam loss with 8 positives + 50 mined negatives ----
    float oc[CCAM], osum = 0.f, m2 = -1e30f;
#pragma unroll
    for (int c = 0; c < CCAM; ++c) {
        oc[c] = oriv[t * CCAM + c] * INV_T;
        osum += oc[c];
        m2 = fmaxf(m2, oc[c]);
    }
    for (int j = 0; j < KTOP; ++j) m2 = fmaxf(m2, my[j] * INV_T);
    float s2 = 0.f;
#pragma unroll
    for (int c = 0; c < CCAM; ++c) s2 += __expf(oc[c] - m2);
    for (int j = 0; j < KTOP; ++j) s2 += __expf(my[j] * INV_T - m2);
    float loss_k = (m2 + __logf(s2)) - osum * (1.0f / CCAM);

    // ---- per-camera segment mean, then sum ----
    __shared__ float si[CCAM], sk[CCAM], cn[CCAM];
    if (t < CCAM) { si[t] = 0.f; sk[t] = 0.f; cn[t] = 0.f; }
    __syncthreads();
    atomicAdd(&si[cam], ce);
    atomicAdd(&sk[cam], loss_k);
    atomicAdd(&cn[cam], 1.0f);
    __syncthreads();
    if (t == 0) {
        float li = 0.f, lk = 0.f;
        for (int c = 0; c < CCAM; ++c) {
            if (cn[c] > 0.f) { li += si[c] / cn[c]; lk += sk[c] / cn[c]; }
        }
        out[0] = li;
        out[1] = 0.5f * lk;                         // LOSS_WEIGHT
    }
}

// ---------------------------------------------------------------------------
extern "C" void kernel_launch(void* const* d_in, const int* in_sizes, int n_in,
                              void* d_out, int out_size, void* d_ws, size_t ws_size,
                              hipStream_t stream) {
    (void)in_sizes; (void)n_in; (void)out_size; (void)ws_size;
    const float* x      = (const float*)d_in[0];
    const int*   labels = (const int*)d_in[1];
    const int*   cams   = (const int*)d_in[2];
    const float* tempV  = (const float*)d_in[3];

    char* ws = (char*)d_ws;
    unsigned* xTb     = (unsigned*)ws;                               // 1 MB
    float*    logitsT = (float*)(ws + (1u << 20));                   // 16 MB
    char*     tail    = ws + (1u << 20) + ((size_t)NBANK * B_SZ * 4);
    float* pmax = (float*)tail;                                      // 32 KB
    float* psum = pmax + NSLICE * B_SZ;                              // 32 KB
    float* oriv = psum + NSLICE * B_SZ;                              // 8 KB
    float* ptop = oriv + B_SZ * CCAM;                                // 1.6 MB
    float* out  = (float*)d_out;

    capNormPack<<<B_SZ, 256, 0, stream>>>(x, xTb);
    capGemm<<<NBANK / 128, 256, 0, stream>>>(tempV, xTb, logitsT);
    capPartials<<<NSLICE, 256, 0, stream>>>(logitsT, labels, cams,
                                            pmax, psum, ptop, oriv);
    capFinal<<<1, 256, 0, stream>>>(pmax, psum, ptop, oriv, cams, out);
}
